// TopKMoeLayer_4999341932688
// MI455X (gfx1250) — compile-verified, matrix operand order checked by reading, not run on hardware
//
#include <hip/hip_runtime.h>
#include <math.h>

// Problem constants (match reference)
constexpr int NB = 2, SQ = 2048, DM = 2048, FF = 8192, EE = 8, KK = 2;
constexpr int NT = NB * SQ;                  // 4096 tokens
constexpr int MT = 128;                      // M tile (rows)
constexpr int ROWCAP = NT * KK + EE * MT;    // padded assignment rows = 9216
constexpr int ROWBLKS = ROWCAP / MT;         // 72

typedef __attribute__((ext_vector_type(16))) __bf16 v16bf;
typedef __attribute__((ext_vector_type(8)))  float  v8f;

union FragU {
    v16bf v;
    uint4 q[2];
};

static __device__ __forceinline__ unsigned short f2bf(float f) {
    unsigned u = __float_as_uint(f);
    unsigned r = u + 0x7FFFu + ((u >> 16) & 1u);   // RNE
    return (unsigned short)(r >> 16);
}
static __device__ __forceinline__ unsigned pack2bf(float lo, float hi) {
    return (unsigned)f2bf(lo) | ((unsigned)f2bf(hi) << 16);
}
static __device__ __forceinline__ float gelu_tanh(float x) {
    float x3 = x * x * x;
    float t = tanhf(0.7978845608028654f * (x + 0.044715f * x3));
    return 0.5f * x * (1.0f + t);
}
static __device__ __forceinline__ v8f wmma_bf16(const FragU& a, const FragU& b, v8f c) {
    return __builtin_amdgcn_wmma_f32_16x16x32_bf16(false, a.v, false, b.v,
                                                   (short)0, c, false, false);
}
// CDNA5 async copy: 16B global -> LDS per lane, tracked by ASYNCcnt.
static __device__ __forceinline__ void async_b128(unsigned lds_off, const void* gaddr) {
    asm volatile("global_load_async_to_lds_b128 %0, %1, off"
                 :: "v"(lds_off), "v"(gaddr) : "memory");
}
static __device__ __forceinline__ void wait_async0() {
    asm volatile("s_wait_asynccnt 0" ::: "memory");
}
static __device__ __forceinline__ unsigned lds_addr(const void* p) {
    // generic address of a __shared__ object: low 32 bits are the LDS byte address
    return (unsigned)(unsigned long long)p;
}

// ---------------- init ----------------
__global__ void k_init(int* counts, int* cursors, int* tok_of_row) {
    int t = blockIdx.x * blockDim.x + threadIdx.x;
    if (t < EE) { counts[t] = 0; cursors[t] = 0; }
    if (t < ROWCAP) tok_of_row[t] = NT;        // padding rows -> zero row of xbf
}

// ---------------- fp32 -> bf16 activation convert (+1 zero row) ----------------
__global__ void k_cvt_x(const float* __restrict__ x, unsigned short* __restrict__ xbf, int n) {
    int t = blockIdx.x * blockDim.x + threadIdx.x;
    if (t < n) xbf[t] = f2bf(x[t]);
    else if (t < n + DM) xbf[t] = 0;           // zero row at token index NT
}

// ---------------- gating: one wave per token ----------------
__global__ void k_gate(const float* __restrict__ x, const float* __restrict__ gw,
                       const float* __restrict__ gb,
                       int* __restrict__ topk_idx, float* __restrict__ topk_w,
                       int* __restrict__ counts) {
    int wave = threadIdx.x >> 5;
    int lane = threadIdx.x & 31;
    int n = blockIdx.x * 8 + wave;
    if (n >= NT) return;

    int e = lane & 7;
    int part = lane >> 3;
    const float* xr = x + (size_t)n * DM;
    float acc = 0.0f;
    for (int d = part * (DM / 4); d < (part + 1) * (DM / 4); ++d)
        acc += xr[d] * gw[d * EE + e];
    acc += __shfl_xor(acc, 8, 32);
    acc += __shfl_xor(acc, 16, 32);
    float logit = acc + gb[e];

    float lg[EE];
#pragma unroll
    for (int i = 0; i < EE; ++i) lg[i] = __shfl(logit, i, 32);

    if (lane == 0) {
        float mx = lg[0];
#pragma unroll
        for (int i = 1; i < EE; ++i) mx = fmaxf(mx, lg[i]);
        float p[EE];
#pragma unroll
        for (int i = 0; i < EE; ++i) p[i] = __expf(lg[i] - mx);
        int i0 = 0;
#pragma unroll
        for (int i = 1; i < EE; ++i) if (p[i] > p[i0]) i0 = i;   // first-on-tie
        int i1 = (i0 == 0) ? 1 : 0;
#pragma unroll
        for (int i = 0; i < EE; ++i) if (i != i0 && p[i] > p[i1]) i1 = i;
        float s = p[i0] + p[i1];
        topk_idx[n * KK + 0] = i0;
        topk_idx[n * KK + 1] = i1;
        topk_w[n * KK + 0] = p[i0] / s;
        topk_w[n * KK + 1] = p[i1] / s;
        atomicAdd(&counts[i0], 1);
        atomicAdd(&counts[i1], 1);
    }
}

// ---------------- padded exclusive scan (1 thread) ----------------
__global__ void k_scan(const int* __restrict__ counts, int* __restrict__ offp,
                       int* __restrict__ cursors) {
    offp[0] = 0;
    for (int e = 0; e < EE; ++e) {
        offp[e + 1] = offp[e] + ((counts[e] + MT - 1) & ~(MT - 1));
        cursors[e] = 0;
    }
}

// ---------------- assignment compaction ----------------
__global__ void k_assign(const int* __restrict__ topk_idx, const int* __restrict__ offp,
                         int* __restrict__ cursors, int* __restrict__ tok_of_row,
                         int* __restrict__ pos_map) {
    int t = blockIdx.x * blockDim.x + threadIdx.x;
    if (t >= NT * KK) return;
    int n = t / KK;
    int e = topk_idx[t];
    int p = atomicAdd(&cursors[e], 1);
    int row = offp[e] + p;
    tok_of_row[row] = n;
    pos_map[t] = row;
}

// ======================= GEMM kernels: 128x128 tiles =======================
// 8 waves; wave (wm = w&3, wn = w>>2) computes a 32x64 subtile = 8 WMMA accums.
// A tile (128x32 bf16) staged by async_b128; B tile (32x128) converted fp32->bf16
// and stored transposed (k contiguous) with packed ds_store_b32.

// ---------------- GEMM1: H = gelu(gather(X) @ w1[e] + b1[e]), bf16 out ----------------
__global__ __launch_bounds__(256)
void k_gemm1(const unsigned short* __restrict__ xbf,
             const float* __restrict__ w1, const float* __restrict__ b1,
             const int* __restrict__ offp, const int* __restrict__ tok_of_row,
             unsigned short* __restrict__ H) {
    __shared__ __align__(16) unsigned short As[MT][40];
    __shared__ __align__(16) unsigned short Bs[MT][40];
    __shared__ int toks[MT];

    int row0 = blockIdx.y * MT;
    if (row0 >= offp[EE]) return;
    int f0 = blockIdx.x * 128;

    int e = 0;
    while (e < EE - 1 && row0 >= offp[e + 1]) ++e;

    int t = threadIdx.x;
    if (t < MT) toks[t] = tok_of_row[row0 + t];
    __syncthreads();

    int wave = t >> 5, lane = t & 31;
    int wm = wave & 3, wn = wave >> 2;
    int half = lane >> 4, nl = lane & 15;

    // A staging: 2 x 16B units per thread
    int am[2], akg[2], atok[2];
    unsigned alds[2];
#pragma unroll
    for (int i = 0; i < 2; ++i) {
        int u = i * 256 + t;
        am[i] = u >> 2; akg[i] = u & 3;
        atok[i] = toks[am[i]];
        alds[i] = lds_addr(&As[am[i]][akg[i] * 8]);
    }
    // B staging: 8 d-pair units per thread
    int bf_[8], bdl[8];
#pragma unroll
    for (int i = 0; i < 8; ++i) {
        int u = i * 256 + t;
        bf_[i] = u & 127; bdl[i] = (u >> 7) * 2;
    }

    const float* w1e = w1 + (size_t)e * DM * FF;
    v8f acc[2][4] = {};

    for (int kk = 0; kk < DM; kk += 32) {
        __syncthreads();
#pragma unroll
        for (int i = 0; i < 2; ++i)
            async_b128(alds[i], xbf + (size_t)atok[i] * DM + kk + akg[i] * 8);
        if (kk + 32 < DM)
            __builtin_prefetch(&w1e[(size_t)(kk + 32 + bdl[0]) * FF + f0 + bf_[0]], 0, 1);
#pragma unroll
        for (int i = 0; i < 8; ++i) {
            const float* src = &w1e[(size_t)(kk + bdl[i]) * FF + f0 + bf_[i]];
            *(unsigned*)&Bs[bf_[i]][bdl[i]] = pack2bf(src[0], src[FF]);
        }
        wait_async0();
        __syncthreads();

        FragU a0, a1;
        int mb = wm * 32 + nl;
        a0.q[0] = *(const uint4*)&As[mb][half << 3];
        a0.q[1] = *(const uint4*)&As[mb][16 + (half << 3)];
        a1.q[0] = *(const uint4*)&As[mb + 16][half << 3];
        a1.q[1] = *(const uint4*)&As[mb + 16][16 + (half << 3)];
#pragma unroll
        for (int j = 0; j < 4; ++j) {
            FragU b;
            int n0 = wn * 64 + j * 16 + nl;
            b.q[0] = *(const uint4*)&Bs[n0][half << 3];
            b.q[1] = *(const uint4*)&Bs[n0][16 + (half << 3)];
            acc[0][j] = wmma_bf16(a0, b, acc[0][j]);
            acc[1][j] = wmma_bf16(a1, b, acc[1][j]);
        }
    }

    const float* b1e = b1 + (size_t)e * FF;
#pragma unroll
    for (int ms = 0; ms < 2; ++ms)
#pragma unroll
        for (int j = 0; j < 4; ++j) {
            v8f& c = acc[ms][j];
            int ng = f0 + wn * 64 + j * 16 + nl;
            float bias = b1e[ng];
#pragma unroll
            for (int r = 0; r < 8; ++r) {
                int row = row0 + wm * 32 + ms * 16 + r + 8 * half;
                H[(size_t)row * FF + ng] = f2bf(gelu_tanh(c[r] + bias));
            }
        }
}

// ---------------- GEMM2: Y = H @ w2[e] + b2[e], f32 out ----------------
__global__ __launch_bounds__(256)
void k_gemm2(const unsigned short* __restrict__ H,
             const float* __restrict__ w2, const float* __restrict__ b2,
             const int* __restrict__ offp, float* __restrict__ Y) {
    __shared__ __align__(16) unsigned short As[MT][40];
    __shared__ __align__(16) unsigned short Bs[MT][40];

    int row0 = blockIdx.y * MT;
    if (row0 >= offp[EE]) return;
    int d0 = blockIdx.x * 128;

    int e = 0;
    while (e < EE - 1 && row0 >= offp[e + 1]) ++e;

    int t = threadIdx.x;
    int wave = t >> 5, lane = t & 31;
    int wm = wave & 3, wn = wave >> 2;
    int half = lane >> 4, nl = lane & 15;

    int am[2], akg[2];
    unsigned alds[2];
#pragma unroll
    for (int i = 0; i < 2; ++i) {
        int u = i * 256 + t;
        am[i] = u >> 2; akg[i] = u & 3;
        alds[i] = lds_addr(&As[am[i]][akg[i] * 8]);
    }
    int bf_[8], bdl[8];
#pragma unroll
    for (int i = 0; i < 8; ++i) {
        int u = i * 256 + t;
        bf_[i] = u & 127; bdl[i] = (u >> 7) * 2;
    }

    const float* w2e = w2 + (size_t)e * FF * DM;
    v8f acc[2][4] = {};

    for (int kk = 0; kk < FF; kk += 32) {
        __syncthreads();
#pragma unroll
        for (int i = 0; i < 2; ++i)
            async_b128(alds[i], H + (size_t)(row0 + am[i]) * FF + kk + akg[i] * 8);
        if (kk + 32 < FF)
            __builtin_prefetch(&w2e[(size_t)(kk + 32 + bdl[0]) * DM + d0 + bf_[0]], 0, 1);
#pragma unroll
        for (int i = 0; i < 8; ++i) {
            const float* src = &w2e[(size_t)(kk + bdl[i]) * DM + d0 + bf_[i]];
            *(unsigned*)&Bs[bf_[i]][bdl[i]] = pack2bf(src[0], src[DM]);
        }
        wait_async0();
        __syncthreads();

        FragU a0, a1;
        int mb = wm * 32 + nl;
        a0.q[0] = *(const uint4*)&As[mb][half << 3];
        a0.q[1] = *(const uint4*)&As[mb][16 + (half << 3)];
        a1.q[0] = *(const uint4*)&As[mb + 16][half << 3];
        a1.q[1] = *(const uint4*)&As[mb + 16][16 + (half << 3)];
#pragma unroll
        for (int j = 0; j < 4; ++j) {
            FragU b;
            int n0 = wn * 64 + j * 16 + nl;
            b.q[0] = *(const uint4*)&Bs[n0][half << 3];
            b.q[1] = *(const uint4*)&Bs[n0][16 + (half << 3)];
            acc[0][j] = wmma_bf16(a0, b, acc[0][j]);
            acc[1][j] = wmma_bf16(a1, b, acc[1][j]);
        }
    }

    const float* b2e = b2 + (size_t)e * DM;
#pragma unroll
    for (int ms = 0; ms < 2; ++ms)
#pragma unroll
        for (int j = 0; j < 4; ++j) {
            v8f& c = acc[ms][j];
            int ng = d0 + wn * 64 + j * 16 + nl;
            float bias = b2e[ng];
#pragma unroll
            for (int r = 0; r < 8; ++r) {
                int row = row0 + wm * 32 + ms * 16 + r + 8 * half;
                Y[(size_t)row * DM + ng] = c[r] + bias;
            }
        }
}

// ---------------- weighted combine (fixed k order => deterministic) ----------------
__global__ void k_combine(const float* __restrict__ Y, const int* __restrict__ pos_map,
                          const float* __restrict__ topk_w, float* __restrict__ out) {
    int t = blockIdx.x * blockDim.x + threadIdx.x;
    if (t >= NT * DM) return;
    int n = t / DM, d = t % DM;
    int r0 = pos_map[n * KK + 0], r1 = pos_map[n * KK + 1];
    float w0 = topk_w[n * KK + 0], w1 = topk_w[n * KK + 1];
    out[t] = w0 * Y[(size_t)r0 * DM + d] + w1 * Y[(size_t)r1 * DM + d];
}

extern "C" void kernel_launch(void* const* d_in, const int* in_sizes, int n_in,
                              void* d_out, int out_size, void* d_ws, size_t ws_size,
                              hipStream_t stream) {
    const float* x   = (const float*)d_in[0];
    const float* gw  = (const float*)d_in[1];
    const float* gb  = (const float*)d_in[2];
    const float* w1  = (const float*)d_in[3];
    const float* b1  = (const float*)d_in[4];
    const float* w2  = (const float*)d_in[5];
    const float* b2  = (const float*)d_in[6];
    float* out = (float*)d_out;

    char* ws = (char*)d_ws;
    size_t o = 0;
    auto alloc = [&](size_t bytes) { size_t r = o; o = (o + bytes + 255) & ~(size_t)255; return r; };
    int*   counts     = (int*)  (ws + alloc(EE * 4));
    int*   cursors    = (int*)  (ws + alloc(EE * 4));
    int*   offp       = (int*)  (ws + alloc((EE + 1) * 4));
    int*   topk_idx   = (int*)  (ws + alloc((size_t)NT * KK * 4));
    float* topk_w     = (float*)(ws + alloc((size_t)NT * KK * 4));
    int*   pos_map    = (int*)  (ws + alloc((size_t)NT * KK * 4));
    int*   tok_of_row = (int*)  (ws + alloc((size_t)ROWCAP * 4));
    unsigned short* xbf = (unsigned short*)(ws + alloc((size_t)(NT + 1) * DM * 2)); // +zero row
    unsigned short* H   = (unsigned short*)(ws + alloc((size_t)ROWCAP * FF * 2));
    float* Y            = (float*)(ws + alloc((size_t)ROWCAP * DM * 4));
    (void)ws_size; (void)in_sizes; (void)n_in; (void)out_size;

    k_init<<<(ROWCAP + 255) / 256, 256, 0, stream>>>(counts, cursors, tok_of_row);
    {
        int n = NT * DM;
        k_cvt_x<<<(n + DM + 255) / 256, 256, 0, stream>>>(x, xbf, n);
    }
    k_gate<<<NT / 8, 256, 0, stream>>>(x, gw, gb, topk_idx, topk_w, counts);
    k_scan<<<1, 1, 0, stream>>>(counts, offp, cursors);
    k_assign<<<(NT * KK + 255) / 256, 256, 0, stream>>>(topk_idx, offp, cursors,
                                                        tok_of_row, pos_map);
    k_gemm1<<<dim3(FF / 128, ROWBLKS), 256, 0, stream>>>(xbf, w1, b1, offp, tok_of_row, H);
    k_gemm2<<<dim3(DM / 128, ROWBLKS), 256, 0, stream>>>(H, w2, b2, offp, Y);
    {
        int n = NT * DM;
        k_combine<<<(n + 255) / 256, 256, 0, stream>>>(Y, pos_map, topk_w, out);
    }
}
